// GroupedQueryAttention_3427383902430
// MI455X (gfx1250) — matrix operand order repaired and checked by reading, not compiled, not run on hardware
//
#include <hip/hip_runtime.h>
#include <hip/hip_bf16.h>
#include <cstdint>

// ---------------------------------------------------------------------------
// GQA forward for MI455X (gfx1250, wave32, WMMA f32<-f16 16x16x32)
//   B=2 S=2048 DIN=2048 H=32 G=8 D=64  -> out(f32) ++ k(f32) ++ v(f32)
// Compute-bound (~223 GFLOP vs ~120MB HBM): all GEMMs + attention via
// v_wmma_f32_16x16x32_f16. GEMM uses async global->LDS double-buffered B
// staging (ASYNCcnt path) with 64x64 block tiles for 4x L2-intensity.
// ---------------------------------------------------------------------------

typedef __attribute__((ext_vector_type(16))) _Float16 v16h;
typedef __attribute__((ext_vector_type(8)))  _Float16 v8h;
typedef __attribute__((ext_vector_type(8)))  float    v8f;

#define B_    2
#define S_    2048
#define DIN_  2048
#define H_    32
#define G_    8
#define D_    64
#define GS_   4           // H/G
#define NQKV  3072        // H*D + G*D + G*D
#define ROWS  (B_*S_)     // 4096

union V16U { v16h v; v8h h[2]; };

static __device__ __forceinline__ v8f vzero8() {
  v8f z = {0.f,0.f,0.f,0.f,0.f,0.f,0.f,0.f};
  return z;
}

static __device__ __forceinline__ v8f wmma16(v16h a, v16h b, v8f c) {
  // (neg_a, A, neg_b, B, c_mod, C, reuse_a, reuse_b)
  return __builtin_amdgcn_wmma_f32_16x16x32_f16(false, a, false, b, (short)0, c, false, false);
}

// CDNA5 async global->LDS copy (16B per lane), tracked by ASYNCcnt.
static __device__ __forceinline__ void async_cp16(uint32_t lds_addr, const _Float16* g) {
  asm volatile("global_load_async_to_lds_b128 %0, %1, off"
               :: "v"(lds_addr), "v"((uint64_t)(uintptr_t)g)
               : "memory");
}

static __device__ __forceinline__ void wait_async0() {
#if __has_builtin(__builtin_amdgcn_s_wait_asynccnt)
  __builtin_amdgcn_s_wait_asynccnt(0);
#else
  asm volatile("s_wait_asynccnt 0x0" ::: "memory");
#endif
}

// Generic (__shared__) pointer -> 32-bit LDS offset for async VDST.
static __device__ __forceinline__ uint32_t lds_offset_of(const void* p) {
  return (uint32_t)(uintptr_t)(__attribute__((address_space(3))) const void*)p;
}

// ---------------------------------------------------------------------------
// f32 -> f16 conversion of activations
__global__ void k_cvt_x(const float* __restrict__ x, _Float16* __restrict__ xh, int n) {
  int i = blockIdx.x * blockDim.x + threadIdx.x;
  if (i < n) xh[i] = (_Float16)x[i];
}

// Build transposed f16 weights: WqkvT[n][k] (n<2048:Wq, <2560:Wk, else Wv), WoT[n][k]
__global__ void k_cvt_w(const float* __restrict__ Wq, const float* __restrict__ Wk,
                        const float* __restrict__ Wv, const float* __restrict__ Wo,
                        _Float16* __restrict__ WqkvT, _Float16* __restrict__ WoT) {
  int i = blockIdx.x * blockDim.x + threadIdx.x;
  const int n1 = NQKV * DIN_;
  if (i < n1) {
    int n = i / DIN_, k = i % DIN_;
    float v;
    if      (n < 2048) v = Wq[(size_t)k * 2048 + n];
    else if (n < 2560) v = Wk[(size_t)k * 512 + (n - 2048)];
    else               v = Wv[(size_t)k * 512 + (n - 2560)];
    WqkvT[i] = (_Float16)v;
  } else {
    int j = i - n1;
    if (j < DIN_ * DIN_) {
      int n = j / DIN_, k = j % DIN_;
      WoT[j] = (_Float16)Wo[(size_t)k * DIN_ + n];
    }
  }
}

// ---------------------------------------------------------------------------
// Block (4 waves) computes 64(M) x 64(N). Wave w owns rows [w*16, w*16+16).
// A-frags direct from global (2x b128/lane, reused across 4 N-subtiles).
// Shared B tile (Bt rows n0..n0+63 x 32 k-halves = 4KB) staged to LDS with
// global_load_async_to_lds_b128, double-buffered; fragments via ds_load_b128.
// Frag layouts per CDNA5 ISA:
//   A (16x32 MxK): lane L: M=L%16; halves0-7 -> K=(L>=16?8:0)+0..7,
//                  halves8-15 -> K=16+(L>=16?8:0)+0..7
//   B (32x16 KxN): lane L: N=L%16; halves0-7 -> K=(L>=16?8:0)+0..7, 8-15 -> K=16+(L>=16?8:0)+0..7 (as A)
//   C (16x16):     lane L, vgpr r: M=r+(L>=16?8:0), N=L%16
#define BM 64
#define BN 64
__global__ void __launch_bounds__(128)
k_gemm(const _Float16* __restrict__ A, const _Float16* __restrict__ Bt,
       float* __restrict__ C, int M, int N, int K) {
  __shared__ alignas(16) _Float16 Blds[2][BN * 32];   // 8 KB double-buffered
  const int tid  = threadIdx.x;
  const int lane = tid & 31;
  const int wave = tid >> 5;
  const int hw = lane >> 4, l16 = lane & 15;
  const int nbt = N / BN;
  const int tm = blockIdx.x / nbt, tn = blockIdx.x % nbt;
  const int n0 = tn * BN;

  // B staging: 64 rows x 64B = 256 chunks of 16B; lane handles chunks tid, tid+128
  const uint32_t lds0 = lds_offset_of(&Blds[0][0]);
  const int r0 = tid >> 2,         c0 = tid & 3;
  const int r1 = (tid + 128) >> 2, c1 = (tid + 128) & 3;
  const _Float16* Bg0 = Bt + (size_t)(n0 + r0) * K + c0 * 8;
  const _Float16* Bg1 = Bt + (size_t)(n0 + r1) * K + c1 * 8;
  const uint32_t lo0 = (uint32_t)(r0 * 64 + c0 * 16);
  const uint32_t lo1 = (uint32_t)(r1 * 64 + c1 * 16);

  const _Float16* Arow = A + (size_t)(tm * BM + wave * 16 + l16) * K;

  v8f acc[4];
#pragma unroll
  for (int j = 0; j < 4; ++j) acc[j] = vzero8();

  // prologue: stage kb=0 into buffer 0
  async_cp16(lds0 + lo0, Bg0);
  async_cp16(lds0 + lo1, Bg1);

  for (int kb = 0; kb < K; kb += 32) {
    const int cur = (kb >> 5) & 1;
    wait_async0();          // own async writes done
    __syncthreads();        // all waves' writes done; all reads of prev done
    if (kb + 32 < K) {      // prefetch next K-chunk into other buffer
      const uint32_t nb = lds0 + (uint32_t)((cur ^ 1) * (BN * 32 * 2));
      async_cp16(nb + lo0, Bg0 + kb + 32);
      async_cp16(nb + lo1, Bg1 + kb + 32);
    }
    V16U a;
    a.h[0] = *(const v8h*)(Arow + kb + hw * 8);
    a.h[1] = *(const v8h*)(Arow + kb + 16 + hw * 8);
    const _Float16* Bl = &Blds[cur][0];
#pragma unroll
    for (int j = 0; j < 4; ++j) {
      V16U b;
      b.h[0] = *(const v8h*)(Bl + (j * 16 + l16) * 32 + hw * 8);
      b.h[1] = *(const v8h*)(Bl + (j * 16 + l16) * 32 + 16 + hw * 8);
      acc[j] = wmma16(a.v, b.v, acc[j]);
    }
  }

  float* Crow = C + (size_t)(tm * BM + wave * 16 + hw * 8) * N + n0 + l16;
#pragma unroll
  for (int j = 0; j < 4; ++j)
#pragma unroll
    for (int r = 0; r < 8; ++r) Crow[(size_t)r * N + j * 16] = acc[j][r];
}

// ---------------------------------------------------------------------------
// RMSNorm(D=64) + RoPE. One wave per (b, s, head) over 48 heads (32 q, 8 k, 8 v).
// Lane t owns d=t and d=t+32 (the RoPE rotation pair).
__global__ void __launch_bounds__(32)
k_normrope(const float* __restrict__ pre, const float* __restrict__ cosb,
           const float* __restrict__ sinb, const float* __restrict__ qs,
           const float* __restrict__ ks,
           _Float16* __restrict__ qh, _Float16* __restrict__ kh,
           _Float16* __restrict__ vT, float* __restrict__ kout,
           float* __restrict__ vout) {
  int id = blockIdx.x;
  int hh = id % 48;
  int s  = (id / 48) % S_;
  int b  = id / (48 * S_);
  int t  = threadIdx.x;
  size_t row = (size_t)(b * S_ + s) * NQKV;
  int colbase = (hh < H_) ? hh * D_
              : (hh < H_ + G_) ? 2048 + (hh - H_) * D_
              : 2560 + (hh - H_ - G_) * D_;
  float a  = pre[row + colbase + t];
  float c2 = pre[row + colbase + t + 32];

  if (hh >= H_ + G_) {                       // v head: plain copy
    int g = hh - H_ - G_;
    size_t vidx = ((size_t)(b * G_ + g) * S_ + s) * D_;
    vout[vidx + t]      = a;
    vout[vidx + t + 32] = c2;
    size_t vtb = (size_t)(b * G_ + g) * D_;  // transposed [d][s] for P*V B-frags
    vT[(vtb + t)      * S_ + s] = (_Float16)a;
    vT[(vtb + t + 32) * S_ + s] = (_Float16)c2;
    return;
  }
  float ss = a * a + c2 * c2;
#pragma unroll
  for (int m = 1; m < 32; m <<= 1) ss += __shfl_xor(ss, m, 32);
  float r = rsqrtf(ss * (1.0f / 64.0f) + 1e-6f);
  const float* sc = (hh < H_) ? qs : ks;
  float an = a  * r * sc[t];
  float cn = c2 * r * sc[t + 32];
  float c0 = cosb[s * D_ + t],      s0 = sinb[s * D_ + t];
  float c1 = cosb[s * D_ + t + 32], s1 = sinb[s * D_ + t + 32];
  float o1 = an * c0 - cn * s0;               // d < 32: x*cos - x[d+32]*sin
  float o2 = cn * c1 + an * s1;               // d >=32: x*cos + x[d-32]*sin
  if (hh < H_) {
    size_t qidx = ((size_t)(b * H_ + hh) * S_ + s) * D_;
    qh[qidx + t]      = (_Float16)o1;
    qh[qidx + t + 32] = (_Float16)o2;
  } else {
    int g = hh - H_;
    size_t kidx = ((size_t)(b * G_ + g) * S_ + s) * D_;
    kout[kidx + t]      = o1;
    kout[kidx + t + 32] = o2;
    kh[kidx + t]        = (_Float16)o1;
    kh[kidx + t + 32]   = (_Float16)o2;
  }
}

// ---------------------------------------------------------------------------
// Flash attention: one wave per (b, h, 16-query tile). Causal, online softmax.
__global__ void __launch_bounds__(32)
k_attn(const _Float16* __restrict__ qh, const _Float16* __restrict__ kh,
       const _Float16* __restrict__ vT, _Float16* __restrict__ ctx) {
  __shared__ alignas(16) _Float16 Plds[16 * 32];   // P tile relayout C->A frag
  int lane = threadIdx.x;
  int hw = lane >> 4, l16 = lane & 15;
  int id = blockIdx.x;
  int qt = id % (S_ / 16);
  int h  = (id / (S_ / 16)) % H_;
  int b  = id / ((S_ / 16) * H_);
  int g  = h / GS_;
  int q0 = qt * 16;

  const _Float16* Qrow = qh + ((size_t)(b * H_ + h) * S_ + q0 + l16) * D_;
  const _Float16* Kb   = kh + (size_t)(b * G_ + g) * S_ * D_;   // [s][d]
  const _Float16* Vb   = vT + (size_t)(b * G_ + g) * D_ * S_;   // [d][s]

  V16U qa[2];                                   // Q 16x64 -> two 16x32 A frags
#pragma unroll
  for (int kc = 0; kc < 2; ++kc) {
    qa[kc].h[0] = *(const v8h*)(Qrow + kc * 32 + hw * 16);
    qa[kc].h[1] = *(const v8h*)(Qrow + kc * 32 + hw * 16 + 8);
  }

  float mstat[8], lstat[8];
#pragma unroll
  for (int r = 0; r < 8; ++r) { mstat[r] = -INFINITY; lstat[r] = 0.f; }
  v8f acc[4];
#pragma unroll
  for (int t2 = 0; t2 < 4; ++t2) acc[t2] = vzero8();

  const float scale = 0.125f;                   // 1/sqrt(64)

  for (int kb = 0; kb <= q0 + 15; kb += 32) {
    // S = Q * K^T for 32 keys (two 16x16 score tiles)
    v8f sc0 = vzero8(), sc1 = vzero8();
#pragma unroll
    for (int kc = 0; kc < 2; ++kc) {
      // K^T B-frag: lane: key n = kb(+16)+l16, d = kc*32 + hw*16 + 0..15 (contig)
      v16h bf0 = *(const v16h*)(Kb + (size_t)(kb + l16) * D_ + kc * 32 + hw * 16);
      sc0 = wmma16(qa[kc].v, bf0, sc0);
      v16h bf1 = *(const v16h*)(Kb + (size_t)(kb + 16 + l16) * D_ + kc * 32 + hw * 16);
      sc1 = wmma16(qa[kc].v, bf1, sc1);
    }
    float p0[8], p1[8], alpha[8];
#pragma unroll
    for (int r = 0; r < 8; ++r) {
      int qi = q0 + r + hw * 8;                 // query row of this C element
      float v0 = sc0[r] * scale; if (kb + l16 > qi)      v0 = -INFINITY;
      float v1 = sc1[r] * scale; if (kb + 16 + l16 > qi) v1 = -INFINITY;
      float lm = fmaxf(v0, v1);
#pragma unroll
      for (int m = 1; m < 16; m <<= 1) lm = fmaxf(lm, __shfl_xor(lm, m, 32));
      float mn = fmaxf(mstat[r], lm);
      alpha[r] = __expf(mstat[r] - mn);
      mstat[r] = mn;
      v0 = __expf(v0 - mn);
      v1 = __expf(v1 - mn);
      float rs = v0 + v1;
#pragma unroll
      for (int m = 1; m < 16; m <<= 1) rs += __shfl_xor(rs, m, 32);
      lstat[r] = lstat[r] * alpha[r] + rs;
      p0[r] = v0; p1[r] = v1;
    }
#pragma unroll
    for (int t2 = 0; t2 < 4; ++t2)
#pragma unroll
      for (int r = 0; r < 8; ++r) acc[t2][r] *= alpha[r];

    // P (16x32) C-layout -> LDS -> A-frag layout (DS ops in-order within wave)
#pragma unroll
    for (int r = 0; r < 8; ++r) {
      Plds[(r + hw * 8) * 32 + l16]      = (_Float16)p0[r];
      Plds[(r + hw * 8) * 32 + 16 + l16] = (_Float16)p1[r];
    }
    V16U pa;
    pa.h[0] = *(const v8h*)(&Plds[l16 * 32 + hw * 16]);
    pa.h[1] = *(const v8h*)(&Plds[l16 * 32 + hw * 16 + 8]);

    // acc += P * V ; V B-frag from vT[d][s]: lane: d = t2*16+l16, keys contig
#pragma unroll
    for (int t2 = 0; t2 < 4; ++t2) {
      v16h vb = *(const v16h*)(Vb + (size_t)(t2 * 16 + l16) * S_ + kb + hw * 16);
      acc[t2] = wmma16(pa.v, vb, acc[t2]);
    }
  }

  // epilogue: ctx[b][s][h][d] f16 for the output projection
#pragma unroll
  for (int r = 0; r < 8; ++r) {
    float inv = 1.0f / lstat[r];
    size_t orow = ((size_t)(b * S_) + q0 + r + hw * 8) * (H_ * D_) + h * D_;
#pragma unroll
    for (int t2 = 0; t2 < 4; ++t2)
      ctx[orow + t2 * 16 + l16] = (_Float16)(acc[t2][r] * inv);
  }
}

// ---------------------------------------------------------------------------
extern "C" void kernel_launch(void* const* d_in, const int* in_sizes, int n_in,
                              void* d_out, int out_size, void* d_ws, size_t ws_size,
                              hipStream_t stream) {
  (void)in_sizes; (void)n_in; (void)out_size; (void)ws_size;
  const float* x    = (const float*)d_in[0];
  // d_in[1] = mask (unused; causal mask applied analytically)
  const float* cosb = (const float*)d_in[2];
  const float* sinb = (const float*)d_in[3];
  const float* Wq   = (const float*)d_in[4];
  const float* Wk   = (const float*)d_in[5];
  const float* Wv   = (const float*)d_in[6];
  const float* Wo   = (const float*)d_in[7];
  const float* qs   = (const float*)d_in[8];
  const float* ks   = (const float*)d_in[9];

  char* ws = (char*)d_ws;
  size_t off = 0;
  auto alloc = [&](size_t bytes) -> void* {
    void* p = ws + off;
    off += (bytes + 255) & ~(size_t)255;
    return p;
  };
  _Float16* xh    = (_Float16*)alloc((size_t)ROWS * DIN_ * 2);        // 16 MB
  _Float16* WqkvT = (_Float16*)alloc((size_t)NQKV * DIN_ * 2);        // 12 MB
  _Float16* WoT   = (_Float16*)alloc((size_t)DIN_ * DIN_ * 2);        //  8 MB
  float*    pre   = (float*)   alloc((size_t)ROWS * NQKV * 4);        // 48 MB
  _Float16* qh    = (_Float16*)alloc((size_t)B_ * H_ * S_ * D_ * 2);  // 16 MB
  _Float16* kh    = (_Float16*)alloc((size_t)B_ * G_ * S_ * D_ * 2);  //  4 MB
  _Float16* vT    = (_Float16*)alloc((size_t)B_ * G_ * S_ * D_ * 2);  //  4 MB
  _Float16* ctx   = (_Float16*)alloc((size_t)ROWS * H_ * D_ * 2);     // 16 MB

  float* out  = (float*)d_out;
  float* kout = out  + (size_t)ROWS * DIN_;
  float* vout = kout + (size_t)B_ * G_ * S_ * D_;

  { int n = ROWS * DIN_;
    k_cvt_x<<<(n + 255) / 256, 256, 0, stream>>>(x, xh, n); }
  { int n = NQKV * DIN_ + DIN_ * DIN_;
    k_cvt_w<<<(n + 255) / 256, 256, 0, stream>>>(Wq, Wk, Wv, Wo, WqkvT, WoT); }

  // qkv = x @ [Wq|Wk|Wv] : M=4096 N=3072 K=2048
  k_gemm<<<(ROWS / BM) * (NQKV / BN), 128, 0, stream>>>(xh, WqkvT, pre, ROWS, NQKV, DIN_);

  // per-head RMSNorm + RoPE; emit q/k/v in attention layouts + k/v f32 outputs
  k_normrope<<<B_ * S_ * 48, 32, 0, stream>>>(pre, cosb, sinb, qs, ks,
                                              qh, kh, vT, kout, vout);

  // causal flash attention
  k_attn<<<B_ * H_ * (S_ / 16), 32, 0, stream>>>(qh, kh, vT, ctx);

  // out = ctx @ Wo : M=4096 N=2048 K=2048 (writes f32 straight into d_out)
  k_gemm<<<(ROWS / BM) * (DIN_ / BN), 128, 0, stream>>>(ctx, WoT, out, ROWS, DIN_, DIN_);
}
